// LatentCube_54357106098406
// MI455X (gfx1250) — matrix operand order repaired and checked by reading, not compiled, not run on hardware
//
#include <hip/hip_runtime.h>
#include <hip/hip_fp16.h>

typedef __attribute__((ext_vector_type(16))) _Float16 v16h;
typedef __attribute__((ext_vector_type(8)))  float    v8f;

#define CUBE 32
#define V    32768          // CUBE^3
#define NF   128
#define FIN  384
#define KTOT 131584
#define B    4
#define KC   128            // K staging chunk
#define NT   128            // N tile per block

union ABPack { uint4 u4[2]; v16h h; };

// ---------------------------------------------------------------------------
// 1) Hypernet GEMM: ks[b,k] = sum_l lat[b,l] * Wk[l,k] + bk[k]
//    Memory-bound on Wk (269 MB). lat cached in LDS; coalesced column reads.
// ---------------------------------------------------------------------------
__global__ __launch_bounds__(256) void hyper_kernel(const float* __restrict__ lat,
                                                    const float* __restrict__ Wk,
                                                    const float* __restrict__ bk,
                                                    float* __restrict__ ks) {
    __shared__ float lat_s[B * 512];
    int tid = threadIdx.x;
    for (int i = tid; i < B * 512; i += 256) lat_s[i] = lat[i];
    __syncthreads();
    int k = blockIdx.x * 256 + tid;       // grid.x == KTOT/256 exactly
    float a0, a1, a2, a3;
    a0 = a1 = a2 = a3 = bk[k];
    #pragma unroll 4
    for (int l = 0; l < 512; ++l) {
        float w = Wk[(size_t)l * KTOT + k];
        a0 = fmaf(lat_s[l],        w, a0);
        a1 = fmaf(lat_s[512 + l],  w, a1);
        a2 = fmaf(lat_s[1024 + l], w, a2);
        a3 = fmaf(lat_s[1536 + l], w, a3);
    }
    ks[k]            = a0;
    ks[KTOT + k]     = a1;
    ks[2 * KTOT + k] = a2;
    ks[3 * KTOT + k] = a3;
}

// ---------------------------------------------------------------------------
// 2) Pack dynamic weights to f16 (per-sample), biases stay f32.
//    Per-b f16 layout: w_in@0 (49152), w_mid@49152 (16384),
//                      w_out@65536 (16384), w_sh@81920 (49152)  -> 131072 halves
//    Per-b f32 bias layout: b_in@0, b_mid@128, b_out@256, b_sh@384 -> 512 floats
// ---------------------------------------------------------------------------
__global__ __launch_bounds__(256) void pack_kernel(const float* __restrict__ ks,
                                                   __half* __restrict__ W16,
                                                   float* __restrict__ biasF) {
    int idx = blockIdx.x * 256 + threadIdx.x;
    if (idx >= B * KTOT) return;
    int b = idx / KTOT;
    int i = idx - b * KTOT;
    float val = ks[idx];
    size_t wb = (size_t)b * 131072;
    int bb = b * 512;
    if      (i < 49152)  W16[wb + i] = __half(val);
    else if (i < 49280)  biasF[bb + (i - 49152)] = val;
    else if (i < 65664)  W16[wb + 49152 + (i - 49280)] = __half(val);
    else if (i < 65792)  biasF[bb + 128 + (i - 65664)] = val;
    else if (i < 82176)  W16[wb + 65536 + (i - 65792)] = __half(val);
    else if (i < 82304)  biasF[bb + 256 + (i - 82176)] = val;
    else if (i < 131456) W16[wb + 81920 + (i - 82304)] = __half(val);
    else                 biasF[bb + 384 + (i - 131456)] = val;
}

// ---------------------------------------------------------------------------
// 3) Seed instance-norm -> out[b,c,v] (same data replicated across b)
// ---------------------------------------------------------------------------
__global__ __launch_bounds__(256) void seed_norm_kernel(const float* __restrict__ seed,
                                                        float* __restrict__ outb) {
    int c = blockIdx.x;
    int tid = threadIdx.x;
    const float* src = seed + (size_t)c * V;
    float s = 0.f, s2 = 0.f;
    for (int v = tid; v < V; v += 256) { float x = src[v]; s += x; s2 += x * x; }
    #pragma unroll
    for (int off = 16; off; off >>= 1) { s += __shfl_down(s, off); s2 += __shfl_down(s2, off); }
    __shared__ float r1[8], r2[8], bcast[2];
    int wave = tid >> 5, lane = tid & 31;
    if (lane == 0) { r1[wave] = s; r2[wave] = s2; }
    __syncthreads();
    if (tid == 0) {
        float ts = 0.f, ts2 = 0.f;
        for (int w = 0; w < 8; ++w) { ts += r1[w]; ts2 += r2[w]; }
        float mean = ts * (1.f / V);
        float var  = ts2 * (1.f / V) - mean * mean;
        bcast[0] = mean;
        bcast[1] = rsqrtf(var + 1e-5f);
    }
    __syncthreads();
    float mean = bcast[0], rstd = bcast[1];
    for (int v = tid; v < V; v += 256) {
        float nv = (src[v] - mean) * rstd;
        #pragma unroll
        for (int b = 0; b < B; ++b)
            outb[((size_t)(b * NF + c)) * V + v] = nv;
    }
}

// ---------------------------------------------------------------------------
// SinSobel depthwise 3x3x3 gradient stencil (zero-padded), g = [-1,0,1]
// ---------------------------------------------------------------------------
__device__ __forceinline__ void sobel3(const float* __restrict__ ch, int v,
                                       float& gx, float& gy, float& gz) {
    int x = v >> 10, y = (v >> 5) & 31, w = v & 31;
    float n[3][3][3];
    #pragma unroll
    for (int i = 0; i < 3; ++i) {
        int xx = x + i - 1;
        #pragma unroll
        for (int j = 0; j < 3; ++j) {
            int yy = y + j - 1;
            #pragma unroll
            for (int k = 0; k < 3; ++k) {
                int ww = w + k - 1;
                bool in = ((unsigned)xx < 32u) & ((unsigned)yy < 32u) & ((unsigned)ww < 32u);
                n[i][j][k] = in ? ch[(xx << 10) | (yy << 5) | ww] : 0.f;
            }
        }
    }
    gx = gy = gz = 0.f;
    #pragma unroll
    for (int j = 0; j < 3; ++j)
        #pragma unroll
        for (int k = 0; k < 3; ++k) gx += n[2][j][k] - n[0][j][k];
    #pragma unroll
    for (int i = 0; i < 3; ++i)
        #pragma unroll
        for (int k = 0; k < 3; ++k) gy += n[i][2][k] - n[i][0][k];
    #pragma unroll
    for (int i = 0; i < 3; ++i)
        #pragma unroll
        for (int j = 0; j < 3; ++j) gz += n[i][j][2] - n[i][j][0];
}

__global__ __launch_bounds__(256) void zero_stats_kernel(float* __restrict__ stats) {
    int i = blockIdx.x * 256 + threadIdx.x;
    if (i < B * FIN * 2) stats[i] = 0.f;
}

// Single conv pass: write RAW f16 gradients y16[(b*384+ch)*V+v] and
// accumulate per-channel (sum, sumsq) into stats via wave-reduced atomics.
__global__ __launch_bounds__(256) void conv_kernel(const float* __restrict__ outb,
                                                   __half* __restrict__ y16,
                                                   float* __restrict__ stats) {
    int bc = blockIdx.y;                       // b*128 + c
    int v  = blockIdx.x * 256 + threadIdx.x;
    const float* ch = outb + (size_t)bc * V;
    float g[3];
    sobel3(ch, v, g[0], g[1], g[2]);
    int b = bc >> 7, c = bc & 127;
    #pragma unroll
    for (int d = 0; d < 3; ++d) {
        int chn = b * FIN + 3 * c + d;
        y16[(size_t)chn * V + v] = __half(g[d]);
    }
    float vals[6] = { g[0], g[0] * g[0], g[1], g[1] * g[1], g[2], g[2] * g[2] };
    int lane = threadIdx.x & 31;
    #pragma unroll
    for (int q = 0; q < 6; ++q) {
        float s = vals[q];
        #pragma unroll
        for (int off = 16; off; off >>= 1) s += __shfl_down(s, off);
        if (lane == 0) {
            int chn = b * FIN + 3 * c + (q >> 1);
            atomicAdd(&stats[(chn << 1) + (q & 1)], s);
        }
    }
}

// (sum, sumsq) -> (mean, rstd) in place
__global__ __launch_bounds__(256) void stats_fin_kernel(float* __restrict__ stats) {
    int chn = blockIdx.x * 256 + threadIdx.x;
    if (chn >= B * FIN) return;
    float mean = stats[chn * 2] * (1.f / V);
    float var  = stats[chn * 2 + 1] * (1.f / V) - mean * mean;
    stats[chn * 2]     = mean;
    stats[chn * 2 + 1] = rsqrtf(var + 1e-5f);
}

// ---------------------------------------------------------------------------
// WMMA GEMM core: acc(128 x 128) += W(128xK,f16) * X(K x V slice, f16)
// Block: 256 thr = 8 waves; wave w owns M rows [16w,16w+16), N tile = 128.
// K staged in KC=128 chunks; LDS swizzled [kblk][n][16k] so a B-fragment is
// 32 contiguous bytes per lane. Optional per-k instance-norm applied during
// staging (NORM): x' = (x - mean[k]) * rstd[k].
// ---------------------------------------------------------------------------
template <int K, bool NORM>
__device__ __forceinline__ void gemm_accum(const __half* __restrict__ Wb,
                                           const __half* __restrict__ Xb,
                                           const float* __restrict__ normb,
                                           int n0, __half* xs, v8f acc[8]) {
    int tid  = threadIdx.x;
    int wave = tid >> 5, lane = tid & 31;
    int g = lane >> 4, ln = lane & 15;
    const __half* wrow = Wb + (size_t)((wave << 4) | ln) * K;
    for (int kc0 = 0; kc0 < K; kc0 += KC) {
        // --- stage X chunk (KC x NT) into swizzled LDS ---
        for (int cch = tid; cch < KC * (NT / 8); cch += 256) {
            int kk = cch >> 4;              // 0..KC-1
            int nc = (cch & 15) << 3;       // 0..NT-8
            int k  = kc0 + kk;
            const __half* src = Xb + (size_t)k * V + n0 + nc;
            uint4 dat = *reinterpret_cast<const uint4*>(src);
            const __half* hp = reinterpret_cast<const __half*>(&dat);
            float mean = 0.f, rstd = 1.f;
            if (NORM) { mean = normb[k * 2]; rstd = normb[k * 2 + 1]; }
            int base = ((kk >> 4) * NT + nc) * 16 + (kk & 15);
            #pragma unroll
            for (int j = 0; j < 8; ++j) {
                __half hv = hp[j];
                if (NORM) hv = __half((float(hp[j]) - mean) * rstd);
                xs[base + j * 16] = hv;
            }
        }
        __syncthreads();
        #pragma unroll
        for (int k0 = 0; k0 < KC; k0 += 32) {
            ABPack aU;
            const __half* wr = wrow + kc0 + k0;
            aU.u4[0] = *reinterpret_cast<const uint4*>(wr + 16 * g);
            aU.u4[1] = *reinterpret_cast<const uint4*>(wr + 16 * g + 8);
            #pragma unroll
            for (int t = 0; t < 8; ++t) {
                ABPack bU;
                const __half* bp = xs + (((k0 >> 4) + g) * NT + (t * 16 + ln)) * 16;
                bU.u4[0] = *reinterpret_cast<const uint4*>(bp);
                bU.u4[1] = *reinterpret_cast<const uint4*>(bp + 8);
                acc[t] = __builtin_amdgcn_wmma_f32_16x16x32_f16(
                    false, aU.h, false, bU.h, (short)0, acc[t], false, false);
            }
        }
        __syncthreads();
    }
}

// h_out = relu(W * X' + bias) -> f16   (X' optionally instance-normalized)
template <int K, bool NORM>
__global__ __launch_bounds__(256) void gemm_relu_kernel(const __half* __restrict__ W16, int wOff,
                                                        const __half* __restrict__ X,
                                                        const float* __restrict__ stats,
                                                        const float* __restrict__ biasF, int bOff,
                                                        __half* __restrict__ H) {
    int b  = blockIdx.y;
    int n0 = blockIdx.x * NT;
    const __half* Wb = W16 + (size_t)b * 131072 + wOff;
    const __half* Xb = X + (size_t)b * K * V;
    const float* normb = NORM ? (stats + b * FIN * 2) : nullptr;
    __shared__ __half xs[KC * NT];
    v8f acc[8];
    #pragma unroll
    for (int t = 0; t < 8; ++t)
        #pragma unroll
        for (int r = 0; r < 8; ++r) acc[t][r] = 0.f;
    gemm_accum<K, NORM>(Wb, Xb, normb, n0, xs, acc);
    int tid = threadIdx.x, wave = tid >> 5, lane = tid & 31;
    int g = lane >> 4, ln = lane & 15;
    #pragma unroll
    for (int t = 0; t < 8; ++t)
        #pragma unroll
        for (int r = 0; r < 8; ++r) {
            int m = wave * 16 + g * 8 + r;
            int n = n0 + t * 16 + ln;
            float val = acc[t][r] + biasF[b * 512 + bOff + m];
            val = fmaxf(val, 0.f);
            H[((size_t)(b * NF + m)) * V + n] = __half(val);
        }
}

// out += 0.1 * (w_out*h2 + w_sh*y_norm + b_out + b_sh)
__global__ __launch_bounds__(256) void gemm_final_kernel(const __half* __restrict__ W16,
                                                         const __half* __restrict__ H2,
                                                         const __half* __restrict__ Y16,
                                                         const float* __restrict__ stats,
                                                         const float* __restrict__ biasF,
                                                         float* __restrict__ outb) {
    int b  = blockIdx.y;
    int n0 = blockIdx.x * NT;
    __shared__ __half xs[KC * NT];
    v8f acc[8];
    #pragma unroll
    for (int t = 0; t < 8; ++t)
        #pragma unroll
        for (int r = 0; r < 8; ++r) acc[t][r] = 0.f;
    gemm_accum<128, false>(W16 + (size_t)b * 131072 + 65536, H2 + (size_t)b * 128 * V,
                           nullptr, n0, xs, acc);
    gemm_accum<384, true>(W16 + (size_t)b * 131072 + 81920, Y16 + (size_t)b * 384 * V,
                          stats + b * FIN * 2, n0, xs, acc);
    int tid = threadIdx.x, wave = tid >> 5, lane = tid & 31;
    int g = lane >> 4, ln = lane & 15;
    #pragma unroll
    for (int t = 0; t < 8; ++t)
        #pragma unroll
        for (int r = 0; r < 8; ++r) {
            int m = wave * 16 + g * 8 + r;
            int n = n0 + t * 16 + ln;
            float bias = biasF[b * 512 + 256 + m] + biasF[b * 512 + 384 + m];
            size_t o = ((size_t)(b * NF + m)) * V + n;
            outb[o] = outb[o] + 0.1f * (acc[t][r] + bias);
        }
}

// ---------------------------------------------------------------------------
// final: d_out[b, v] = mean_c out[b, c, v]
// ---------------------------------------------------------------------------
__global__ __launch_bounds__(256) void mean_kernel(const float* __restrict__ outb,
                                                   float* __restrict__ dst, int n) {
    int idx = blockIdx.x * 256 + threadIdx.x;
    if (idx >= n) return;
    int b = idx >> 15;
    int v = idx & (V - 1);
    const float* src = outb + ((size_t)b * NF) * V + v;
    float s = 0.f;
    #pragma unroll 4
    for (int c = 0; c < NF; ++c) s += src[(size_t)c * V];
    dst[idx] = s * (1.f / NF);
}

extern "C" void kernel_launch(void* const* d_in, const int* in_sizes, int n_in,
                              void* d_out, int out_size, void* d_ws, size_t ws_size,
                              hipStream_t stream) {
    const float* lat  = (const float*)d_in[0];
    const float* seed = (const float*)d_in[1];
    const float* Wk   = (const float*)d_in[2];
    const float* bk   = (const float*)d_in[3];
    float* dst = (float*)d_out;

    char* p = (char*)d_ws;
    float*  outb  = (float*)p;  p += (size_t)B * NF * V * 4;     // 67.1 MB
    __half* y16   = (__half*)p; p += (size_t)B * FIN * V * 2;    // 100.7 MB (raw conv)
    __half* h1    = (__half*)p; p += (size_t)B * NF * V * 2;     // 33.6 MB
    __half* h2    = (__half*)p; p += (size_t)B * NF * V * 2;     // 33.6 MB
    float*  ks    = (float*)p;  p += (size_t)B * KTOT * 4;       // 2.1 MB
    __half* W16   = (__half*)p; p += (size_t)B * 131072 * 2;     // 1.0 MB
    float*  biasF = (float*)p;  p += (size_t)B * 512 * 4;
    float*  stats = (float*)p;  p += (size_t)B * FIN * 2 * 4;

    hyper_kernel<<<KTOT / 256, 256, 0, stream>>>(lat, Wk, bk, ks);
    pack_kernel<<<(B * KTOT + 255) / 256, 256, 0, stream>>>(ks, W16, biasF);
    seed_norm_kernel<<<NF, 256, 0, stream>>>(seed, outb);

    for (int it = 0; it < 4; ++it) {
        zero_stats_kernel<<<(B * FIN * 2 + 255) / 256, 256, 0, stream>>>(stats);
        conv_kernel<<<dim3(V / 256, B * NF), 256, 0, stream>>>(outb, y16, stats);
        stats_fin_kernel<<<(B * FIN + 255) / 256, 256, 0, stream>>>(stats);
        gemm_relu_kernel<384, true><<<dim3(V / NT, B), 256, 0, stream>>>(
            W16, 0, y16, stats, biasF, 0, h1);
        gemm_relu_kernel<128, false><<<dim3(V / NT, B), 256, 0, stream>>>(
            W16, 49152, h1, nullptr, biasF, 128, h2);
        gemm_final_kernel<<<dim3(V / NT, B), 256, 0, stream>>>(
            W16, h2, y16, stats, biasF, outb);
    }
    mean_kernel<<<(out_size + 255) / 256, 256, 0, stream>>>(outb, dst, out_size);
}
